// GatingNetwork_15539191677317
// MI455X (gfx1250) — compile-verified, matrix operand order checked by reading, not run on hardware
//
#include <hip/hip_runtime.h>
#include <hip/hip_fp16.h>

typedef __attribute__((ext_vector_type(16))) _Float16 v16h;
typedef __attribute__((ext_vector_type(8)))  _Float16 v8h;
typedef __attribute__((ext_vector_type(8)))  float    v8f;

#define TOKENS 16384
#define DIM    2048
#define EXPERTS 64
#define WAVES_PER_BLOCK 4
#define MTILES 2                                   // two 16-token A tiles per wave
#define TOKENS_PER_WAVE (16 * MTILES)              // 32
#define BLOCKS (TOKENS / (WAVES_PER_BLOCK * TOKENS_PER_WAVE))   // 128

// ---------------------------------------------------------------------------
// helpers
// ---------------------------------------------------------------------------
__device__ inline v16h cvt16(const float4& a0, const float4& a1,
                             const float4& a2, const float4& a3) {
  v16h r;
  r[0]  = (_Float16)a0.x; r[1]  = (_Float16)a0.y; r[2]  = (_Float16)a0.z; r[3]  = (_Float16)a0.w;
  r[4]  = (_Float16)a1.x; r[5]  = (_Float16)a1.y; r[6]  = (_Float16)a1.z; r[7]  = (_Float16)a1.w;
  r[8]  = (_Float16)a2.x; r[9]  = (_Float16)a2.y; r[10] = (_Float16)a2.z; r[11] = (_Float16)a2.w;
  r[12] = (_Float16)a3.x; r[13] = (_Float16)a3.y; r[14] = (_Float16)a3.z; r[15] = (_Float16)a3.w;
  return r;
}

// A-operand tile (16x32 f16 WMMA layout) for token rows starting at base_row.
__device__ inline v16h load_a_tile(const float* __restrict__ xrow, int k0, int hi) {
  const float4 a0 = *(const float4*)(xrow + k0 + hi * 8);
  const float4 a1 = *(const float4*)(xrow + k0 + hi * 8 + 4);
  const float4 a2 = *(const float4*)(xrow + k0 + 16 + hi * 8);
  const float4 a3 = *(const float4*)(xrow + k0 + 16 + hi * 8 + 4);
  return cvt16(a0, a1, a2, a3);
}

__device__ inline bool better(float v, int i, float rv, int ri) {
  // strict "is (v,i) ranked above (rv,ri)" with lower-index tie-break (JAX top_k)
  return (v > rv) || (v == rv && i < ri);
}

// ---------------------------------------------------------------------------
// kernel 0a: zero the p / f accumulators (128 floats in workspace)
// ---------------------------------------------------------------------------
__global__ void moe_init_kernel(float* __restrict__ acc) {
  int i = threadIdx.x;
  if (i < 2 * EXPERTS) acc[i] = 0.0f;
}

// ---------------------------------------------------------------------------
// kernel 0b: one-time W f32 -> f16 conversion (W then lives in L2 as 256KB)
// ---------------------------------------------------------------------------
__global__ void moe_wcvt_kernel(const float* __restrict__ W,
                                _Float16* __restrict__ Wh) {
  const int i = (blockIdx.x * blockDim.x + threadIdx.x) * 4;
  const float4 w = *(const float4*)(W + i);
  v8h* dst2 = nullptr; (void)dst2;
  Wh[i + 0] = (_Float16)w.x;
  Wh[i + 1] = (_Float16)w.y;
  Wh[i + 2] = (_Float16)w.z;
  Wh[i + 3] = (_Float16)w.w;
}

// ---------------------------------------------------------------------------
// shared epilogue: top-2 + softmax + load-balance accumulation for one
// 16-token x 64-expert C tile (4 accumulators).
// ---------------------------------------------------------------------------
__device__ inline void epilogue(const v8f* acc, int token_base, int row, int hi,
                                float* __restrict__ out_gates,
                                float* __restrict__ out_idx,
                                float* p_local, float* f_local) {
#pragma unroll
  for (int j = 0; j < 8; ++j) {
    const int token = token_base + j + 8 * hi;

    float v[4];
    int   e[4];
#pragma unroll
    for (int n = 0; n < 4; ++n) { v[n] = acc[n][j]; e[n] = n * 16 + row; }

    // local top-2 among this lane's 4 experts
    float m0, m1; int i0, i1;
    if (better(v[1], e[1], v[0], e[0])) { m0 = v[1]; i0 = e[1]; m1 = v[0]; i1 = e[0]; }
    else                                { m0 = v[0]; i0 = e[0]; m1 = v[1]; i1 = e[1]; }
#pragma unroll
    for (int n = 2; n < 4; ++n) {
      if (better(v[n], e[n], m0, i0)) { m1 = m0; i1 = i0; m0 = v[n]; i0 = e[n]; }
      else if (better(v[n], e[n], m1, i1)) { m1 = v[n]; i1 = e[n]; }
    }

    // butterfly merge of top-2 across the 16-lane group (xor masks stay in-half)
#pragma unroll
    for (int m = 1; m <= 8; m <<= 1) {
      const float o0 = __shfl_xor(m0, m, 32);
      const float o1 = __shfl_xor(m1, m, 32);
      const int  oi0 = __shfl_xor(i0, m, 32);
      const int  oi1 = __shfl_xor(i1, m, 32);
      if (better(o0, oi0, m0, i0)) {
        const float lv = m0; const int li = i0;          // loser's first
        m0 = o0; i0 = oi0;
        if (better(o1, oi1, lv, li)) { m1 = o1; i1 = oi1; }
        else                         { m1 = lv; i1 = li;  }
      } else if (better(o0, oi0, m1, i1)) {
        m1 = o0; i1 = oi0;
      }
    }
    // all 16 lanes of the group now agree on (m0,i0,m1,i1); m0 = max over 64

    // full 64-wide softmax denominator (for load-balance probs)
    float ex[4];
    float s = 0.0f;
#pragma unroll
    for (int n = 0; n < 4; ++n) { ex[n] = __expf(v[n] - m0); s += ex[n]; }
#pragma unroll
    for (int m = 1; m <= 8; m <<= 1) s += __shfl_xor(s, m, 32);
    const float inv_s = 1.0f / s;

#pragma unroll
    for (int n = 0; n < 4; ++n)
      atomicAdd(&p_local[e[n]], ex[n] * inv_s);         // ds_add_f32

    if (row == 0) {                                      // lanes 0 / 16: one per token
      atomicAdd(&f_local[i0], 1.0f);                     // argmax one-hot count
      const float t  = __expf(m1 - m0);                  // softmax over the pair
      const float g0 = 1.0f / (1.0f + t);
      out_gates[2 * token]     = g0;
      out_gates[2 * token + 1] = t * g0;
      out_idx[2 * token]       = (float)i0;
      out_idx[2 * token + 1]   = (float)i1;
    }
  }
}

// ---------------------------------------------------------------------------
// kernel 1: fused router. One wave handles 32 tokens x 64 experts
// (two M tiles sharing each B load). WF16: B pre-converted to f16.
// ---------------------------------------------------------------------------
template <bool WF16>
__global__ __launch_bounds__(WAVES_PER_BLOCK * 32)
void moe_gate_kernel(const float* __restrict__ x,
                     const void* __restrict__ Wv,
                     float* __restrict__ out_gates,   // [T*2]
                     float* __restrict__ out_idx,     // [T*2], index as float
                     float* __restrict__ p_accum,     // [64] global
                     float* __restrict__ f_accum) {   // [64] global
  __shared__ float p_local[EXPERTS];
  __shared__ float f_local[EXPERTS];

  const int tid = threadIdx.x;
  if (tid < EXPERTS) { p_local[tid] = 0.0f; f_local[tid] = 0.0f; }
  __syncthreads();

  const int wave = tid >> 5;
  const int lane = tid & 31;
  const int row  = lane & 15;   // A: token row | B: expert row | C: column N
  const int hi   = lane >> 4;   // K half / M half of C
  const int token_base = (blockIdx.x * WAVES_PER_BLOCK + wave) * TOKENS_PER_WAVE;

  const float* __restrict__ xrow0 = x + (size_t)(token_base + row) * DIM;
  const float* __restrict__ xrow1 = xrow0 + (size_t)16 * DIM;

  v8f acc0[4] = {};   // tokens [base, base+16)
  v8f acc1[4] = {};   // tokens [base+16, base+32)

  for (int k0 = 0; k0 < DIM; k0 += 32) {
    // prefetch the x stream ~2KB ahead (global_prefetch_b8, speculative RT)
    __builtin_prefetch(xrow0 + k0 + 512, 0, 0);
    __builtin_prefetch(xrow1 + k0 + 512, 0, 0);

    const v16h A0 = load_a_tile(xrow0, k0, hi);
    const v16h A1 = load_a_tile(xrow1, k0, hi);

#pragma unroll
    for (int n = 0; n < 4; ++n) {
      // B tile = W^T tile: W row-major [E, D] loads with A-operand addressing
      v16h B;
      if constexpr (WF16) {
        const _Float16* wrow = (const _Float16*)Wv + (size_t)(n * 16 + row) * DIM + k0;
        const v8h blo = *(const v8h*)(wrow + hi * 8);
        const v8h bhi = *(const v8h*)(wrow + 16 + hi * 8);
        B = __builtin_shufflevector(blo, bhi, 0, 1, 2, 3, 4, 5, 6, 7,
                                    8, 9, 10, 11, 12, 13, 14, 15);
      } else {
        const float* wrow = (const float*)Wv + (size_t)(n * 16 + row) * DIM + k0;
        const float4 b0 = *(const float4*)(wrow + hi * 8);
        const float4 b1 = *(const float4*)(wrow + hi * 8 + 4);
        const float4 b2 = *(const float4*)(wrow + 16 + hi * 8);
        const float4 b3 = *(const float4*)(wrow + 16 + hi * 8 + 4);
        B = cvt16(b0, b1, b2, b3);
      }

      acc0[n] = __builtin_amdgcn_wmma_f32_16x16x32_f16(
          false, A0, false, B, (short)0, acc0[n], false, false);
      acc1[n] = __builtin_amdgcn_wmma_f32_16x16x32_f16(
          false, A1, false, B, (short)0, acc1[n], false, false);
    }
  }

  epilogue(acc0, token_base,      row, hi, out_gates, out_idx, p_local, f_local);
  epilogue(acc1, token_base + 16, row, hi, out_gates, out_idx, p_local, f_local);

  __syncthreads();
  if (tid < EXPERTS) {
    atomicAdd(&p_accum[tid], p_local[tid]);
    atomicAdd(&f_accum[tid], f_local[tid]);
  }
}

// ---------------------------------------------------------------------------
// kernel 2: scalar load-balance loss = E * sum_e (f_e/T) * (p_e/T)
// ---------------------------------------------------------------------------
__global__ void moe_loss_kernel(const float* __restrict__ p_accum,
                                const float* __restrict__ f_accum,
                                float* __restrict__ out_loss) {
  __shared__ float s[EXPERTS];
  const int t = threadIdx.x;
  const float invT = 1.0f / (float)TOKENS;
  s[t] = (f_accum[t] * invT) * (p_accum[t] * invT);
  __syncthreads();
  if (t == 0) {
    float sum = 0.0f;
#pragma unroll
    for (int i = 0; i < EXPERTS; ++i) sum += s[i];
    *out_loss = (float)EXPERTS * sum;
  }
}

// ---------------------------------------------------------------------------
extern "C" void kernel_launch(void* const* d_in, const int* in_sizes, int n_in,
                              void* d_out, int out_size, void* d_ws, size_t ws_size,
                              hipStream_t stream) {
  (void)in_sizes; (void)n_in; (void)out_size;

  const float* x = (const float*)d_in[0];   // [16384, 2048]
  const float* W = (const float*)d_in[1];   // [64, 2048]

  float* out   = (float*)d_out;
  float* gates = out;                 // [T*2]
  float* idxs  = out + 2 * TOKENS;    // [T*2] (indices stored as float values)
  float* loss  = out + 4 * TOKENS;    // [1]

  float*     p_acc = (float*)d_ws;          // [64]
  float*     f_acc = p_acc + EXPERTS;       // [64]
  _Float16*  Wh    = (_Float16*)((char*)d_ws + 512);

  const bool wf16 =
      ws_size >= 512 + (size_t)EXPERTS * DIM * sizeof(_Float16);

  moe_init_kernel<<<1, 128, 0, stream>>>(p_acc);

  if (wf16) {
    // 64*2048 elements, 4 per thread
    moe_wcvt_kernel<<<(EXPERTS * DIM) / (256 * 4), 256, 0, stream>>>(W, Wh);
    moe_gate_kernel<true><<<BLOCKS, WAVES_PER_BLOCK * 32, 0, stream>>>(
        x, (const void*)Wh, gates, idxs, p_acc, f_acc);
  } else {
    moe_gate_kernel<false><<<BLOCKS, WAVES_PER_BLOCK * 32, 0, stream>>>(
        x, (const void*)W, gates, idxs, p_acc, f_acc);
  }

  moe_loss_kernel<<<1, EXPERTS, 0, stream>>>(p_acc, f_acc, loss);
}
